// RevRNN_23545010717137
// MI455X (gfx1250) — matrix operand order repaired and checked by reading, not compiled, not run on hardware
//
#include <hip/hip_runtime.h>
#include <hip/hip_bf16.h>
#include <math.h>

// Sizes fixed by the reference
#define B_   64
#define S_   256
#define I_   512
#define D_   512
#define KW_  1024          // D_ + I_
#define EPSF 1e-5f

typedef __attribute__((ext_vector_type(16))) __bf16 v16bf;
typedef __attribute__((ext_vector_type(8)))  float  v8f;

__device__ __forceinline__ float mish_act(float x) {
    float sp = (x > 20.0f) ? x : log1pf(expf(x));
    return x * tanhf(sp);
}

__device__ __forceinline__ v8f wmma_bf16(const v16bf& a, const v16bf& b, const v8f& c) {
    return __builtin_amdgcn_wmma_f32_16x16x32_bf16(false, a, false, b, (short)0, c,
                                                   false, false);
}

// B-fragment: 16 contiguous bf16 from one weight row (per-lane)
__device__ __forceinline__ void load_bfrag(v16bf& f, const __bf16* __restrict__ p) {
#pragma unroll
    for (int i = 0; i < 16; ++i) f[i] = p[i];
}

// A-fragment: 16 contiguous bf16 per lane (k = 16*(lane/16) + s)
__device__ __forceinline__ void load_afrag(v16bf& f, const __bf16* __restrict__ p) {
#pragma unroll
    for (int i = 0; i < 8; ++i) { f[i] = p[i]; f[8 + i] = p[8 + i]; }
}

// ---------------------------------------------------------------------------
// Weights fp32 -> bf16, natural [out][k] layout (WMMA-B friendly)
// ---------------------------------------------------------------------------
__global__ void revrnn_convert_weights(const float* __restrict__ w0,
                                       const float* __restrict__ w1,
                                       __bf16* __restrict__ w0b,
                                       __bf16* __restrict__ w1b, int n) {
    int i = blockIdx.x * blockDim.x + threadIdx.x;
    if (i < n) {
        w0b[i] = (__bf16)w0[i];
        w1b[i] = (__bf16)w1[i];
    }
}

// ---------------------------------------------------------------------------
// fn_input (B,S,I) fp32 -> Xbf (S*B, I) bf16, s-major rows r = s*64+b
// ---------------------------------------------------------------------------
__global__ void revrnn_convert_x(const float* __restrict__ x,
                                 __bf16* __restrict__ Xbf) {
    int idx = blockIdx.x * blockDim.x + threadIdx.x;   // 0 .. S*B*I-1
    int i = idx & (I_ - 1);
    int r = idx >> 9;            // s*64 + b
    int s = r >> 6, b = r & 63;
    Xbf[idx] = (__bf16)x[((size_t)b * S_ + s) * I_ + i];
}

// ---------------------------------------------------------------------------
// Broadcast hidden_state into the two state halves
// ---------------------------------------------------------------------------
__global__ void revrnn_init_state(const float* __restrict__ hidden,
                                  float* __restrict__ a, float* __restrict__ b) {
    int i = blockIdx.x * blockDim.x + threadIdx.x;     // 0 .. B_*D_-1
    int d = i & (D_ - 1);
    a[i] = hidden[d];
    b[i] = hidden[D_ + d];
}

// ---------------------------------------------------------------------------
// Parallel precompute: Xp[r][d] = sum_i Xbf[r][i] * W[d][D_+i],  r = s*64+b.
// Wave -> 16x64 slab (4 N-tiles).  8192 waves; cross-wave latency hiding.
// ---------------------------------------------------------------------------
__global__ void revrnn_xproj(const __bf16* __restrict__ Xbf,
                             const __bf16* __restrict__ Wb,
                             float* __restrict__ Xp) {
    const int lane = threadIdx.x & 31;
    const int wave = threadIdx.x >> 5;
    const int wid  = blockIdx.x * 8 + wave;   // 0..8191
    const int mt   = wid >> 3;                // row band (16 rows of s*64+b)
    const int n0   = (wid & 7) * 64;          // col group

    const __bf16* arow = Xbf + (size_t)(mt * 16 + (lane & 15)) * I_;
    const int koffA = (lane >> 4) * 16;
    const int koffB = (lane >> 4) * 16;
    const int nl = lane & 15;
    const __bf16* w0r = Wb + (size_t)(n0 +  0 + nl) * KW_ + D_;  // x-part rows
    const __bf16* w1r = Wb + (size_t)(n0 + 16 + nl) * KW_ + D_;
    const __bf16* w2r = Wb + (size_t)(n0 + 32 + nl) * KW_ + D_;
    const __bf16* w3r = Wb + (size_t)(n0 + 48 + nl) * KW_ + D_;

    v8f acc0 = {}, acc1 = {}, acc2 = {}, acc3 = {};
    for (int kt = 0; kt < I_ / 32; ++kt) {
        const int k0 = kt * 32;
        __builtin_prefetch(w0r + k0 + 64, 0, 1);
        v16bf afr, b0, b1, b2, b3;
        load_afrag(afr, arow + k0 + koffA);
        load_bfrag(b0, w0r + k0 + koffB);
        load_bfrag(b1, w1r + k0 + koffB);
        load_bfrag(b2, w2r + k0 + koffB);
        load_bfrag(b3, w3r + k0 + koffB);
        acc0 = wmma_bf16(afr, b0, acc0);
        acc1 = wmma_bf16(afr, b1, acc1);
        acc2 = wmma_bf16(afr, b2, acc2);
        acc3 = wmma_bf16(afr, b3, acc3);
    }
#pragma unroll
    for (int j = 0; j < 8; ++j) {
        const int m = mt * 16 + j + ((lane >> 4) * 8);
        const size_t o = (size_t)m * D_ + n0 + nl;
        Xp[o]      = acc0[j];
        Xp[o + 16] = acc1[j];
        Xp[o + 32] = acc2[j];
        Xp[o + 48] = acc3[j];
    }
}

// ---------------------------------------------------------------------------
// Fused half-step (the sequential critical path):
//   dst[b][n] += mish(norm(src))[b][:] @ Wh[n][:]  +  Xp[b][n]
//   phase 1: every WG redundantly reduces mean / sample-var of all of src
//   phase 2: WG normalizes + Mish its 16-row band into LDS (bf16)
//   phase 3: wave -> 16x32 slab (2 N-tiles), A from LDS, B double-buffered
//            from L2 so WMMAs overlap in-flight loads (per-wave ILP: only
//            64 waves run machine-wide here, no cross-wave hiding).
// Grid: 8 WGs x 256 threads  (4 row bands x 2 N-halves, 8 waves each)
// ---------------------------------------------------------------------------
__global__ void revrnn_halfstep(const float* __restrict__ src,
                                const float* __restrict__ bnw,
                                const float* __restrict__ bnb,
                                const __bf16* __restrict__ Wb,
                                const float* __restrict__ Xp,
                                float* __restrict__ dst) {
    __shared__ float  s_sum[256];
    __shared__ float  s_sq[256];
    __shared__ float  s_stats[2];
    __shared__ __bf16 sA[16 * D_];          // 16 KB: this WG's normalized band

    const int tid  = threadIdx.x;
    const int lane = tid & 31;
    const int wave = tid >> 5;
    const int m0   = (blockIdx.x & 3) * 16;         // row band
    const int nh   = (blockIdx.x >> 2) * 256;       // N-half

    // ---- phase 1: global mean / sample variance (redundant per WG) ----
    {
        float sum = 0.f, sq = 0.f;
        const float4* src4 = (const float4*)src;
#pragma unroll 4
        for (int t = 0; t < (B_ * D_ / 4) / 256; ++t) {   // 32 iters
            float4 v = src4[tid + t * 256];
            sum += v.x + v.y + v.z + v.w;
            sq  += v.x * v.x + v.y * v.y + v.z * v.z + v.w * v.w;
        }
        s_sum[tid] = sum; s_sq[tid] = sq;
    }
    __syncthreads();
    for (int off = 128; off > 0; off >>= 1) {
        if (tid < off) { s_sum[tid] += s_sum[tid + off]; s_sq[tid] += s_sq[tid + off]; }
        __syncthreads();
    }
    if (tid == 0) {
        const float N = (float)(B_ * D_);
        float mean = s_sum[0] / N;
        float var  = (s_sq[0] - N * mean * mean) / (N - 1.0f);
        var = fmaxf(var, 0.0f);
        s_stats[0] = mean;
        s_stats[1] = 1.0f / sqrtf(sqrtf(var) + EPSF);  // ref: /sqrt(std+eps)
    }
    __syncthreads();
    const float mean = s_stats[0], scl = s_stats[1];

    // ---- phase 2: normalize + Mish this WG's band into LDS ----
    for (int t = 0; t < (16 * D_) / 256; ++t) {           // 32 iters
        int idx = tid + t * 256;
        int row = idx >> 9, d = idx & (D_ - 1);
        float v = (src[(size_t)(m0 + row) * D_ + d] - mean) * scl * bnw[d] + bnb[d];
        sA[idx] = (__bf16)mish_act(v);
    }
    __syncthreads();

    // ---- phase 3: WMMA GEMM, wave -> 16x32, double-buffered B ----
    const int n0 = nh + wave * 32;
    const __bf16* arow = sA + (size_t)(lane & 15) * D_;
    const int koffA = (lane >> 4) * 16;
    const int koffB = (lane >> 4) * 16;
    const int nl = lane & 15;
    const __bf16* w0r = Wb + (size_t)(n0 +  0 + nl) * KW_ + koffB;  // h-part rows
    const __bf16* w1r = Wb + (size_t)(n0 + 16 + nl) * KW_ + koffB;

    v8f acc0 = {}, acc1 = {};
    v16bf cb0, cb1, nb0, nb1;
    load_bfrag(cb0, w0r);
    load_bfrag(cb1, w1r);
#pragma unroll
    for (int kt = 0; kt < D_ / 32; ++kt) {
        const int k0 = kt * 32;
        if (kt < D_ / 32 - 1) {               // prefetch next k-tile's B
            load_bfrag(nb0, w0r + k0 + 32);
            load_bfrag(nb1, w1r + k0 + 32);
        }
        v16bf afr;
        load_afrag(afr, arow + k0 + koffA);   // LDS: short latency
        acc0 = wmma_bf16(afr, cb0, acc0);
        acc1 = wmma_bf16(afr, cb1, acc1);
        cb0 = nb0; cb1 = nb1;                 // renamed away by full unroll
    }

    // ---- epilogue: residual + precomputed input projection ----
#pragma unroll
    for (int j = 0; j < 8; ++j) {
        const int m = m0 + j + ((lane >> 4) * 8);
        const size_t o = (size_t)m * D_ + n0 + nl;
        dst[o]      = dst[o]      + acc0[j] + Xp[o];
        dst[o + 16] = dst[o + 16] + acc1[j] + Xp[o + 16];
    }
}

// ---------------------------------------------------------------------------
// Pack final state: out[:, :D] = a (y0), out[:, D:] = b (y1)
// ---------------------------------------------------------------------------
__global__ void revrnn_finalize(const float* __restrict__ a,
                                const float* __restrict__ b,
                                float* __restrict__ out) {
    int i = blockIdx.x * blockDim.x + threadIdx.x;   // 0 .. B_*2D_-1
    int bb = i >> 10;
    int d  = i & 1023;
    out[i] = (d < D_) ? a[bb * D_ + d] : b[bb * D_ + (d - D_)];
}

extern "C" void kernel_launch(void* const* d_in, const int* in_sizes, int n_in,
                              void* d_out, int out_size, void* d_ws, size_t ws_size,
                              hipStream_t stream) {
    (void)in_sizes; (void)n_in; (void)out_size; (void)ws_size;

    const float* x      = (const float*)d_in[0];   // (64,256,512)
    const float* hidden = (const float*)d_in[1];   // (1024)
    const float* W0     = (const float*)d_in[2];   // (512,1024)
    const float* W1     = (const float*)d_in[3];   // (512,1024)
    const float* bw0    = (const float*)d_in[4];
    const float* bb0    = (const float*)d_in[5];
    const float* bw1    = (const float*)d_in[6];
    const float* bb1    = (const float*)d_in[7];
    float* out = (float*)d_out;

    // Workspace carve-up (~82.3 MB total)
    char* ws = (char*)d_ws;
    __bf16* W0b  = (__bf16*)ws; ws += (size_t)D_ * KW_ * sizeof(__bf16);       // 1 MB
    __bf16* W1b  = (__bf16*)ws; ws += (size_t)D_ * KW_ * sizeof(__bf16);       // 1 MB
    __bf16* Xbf  = (__bf16*)ws; ws += (size_t)S_ * B_ * I_ * sizeof(__bf16);   // 16 MB
    float*  Xp0  = (float*)ws;  ws += (size_t)S_ * B_ * D_ * sizeof(float);    // 32 MB
    float*  Xp1  = (float*)ws;  ws += (size_t)S_ * B_ * D_ * sizeof(float);    // 32 MB
    float*  abuf = (float*)ws;  ws += (size_t)B_ * D_ * sizeof(float);         // 128 KB
    float*  bbuf = (float*)ws;  ws += (size_t)B_ * D_ * sizeof(float);         // 128 KB

    // 1) weights -> bf16
    {
        int n = D_ * KW_;
        revrnn_convert_weights<<<(n + 255) / 256, 256, 0, stream>>>(W0, W1, W0b, W1b, n);
    }
    // 2) fn_input -> bf16, transposed to s-major rows
    revrnn_convert_x<<<(S_ * B_ * I_) / 256, 256, 0, stream>>>(x, Xbf);

    // 3) broadcast initial hidden state
    revrnn_init_state<<<(B_ * D_) / 256, 256, 0, stream>>>(hidden, abuf, bbuf);

    // 4) parallel precompute of the input projections for all timesteps
    revrnn_xproj<<<1024, 256, 0, stream>>>(Xbf, W0b, Xp0);
    revrnn_xproj<<<1024, 256, 0, stream>>>(Xbf, W1b, Xp1);

    // 5) sequential recurrence: 256 steps x 2 fused half-steps
    for (int s = 0; s < S_; ++s) {
        const size_t off = (size_t)s * B_ * D_;
        // b += calc0(a) + Xp0[s]   (b becomes y1)
        revrnn_halfstep<<<8, 256, 0, stream>>>(abuf, bw0, bb0, W0b, Xp0 + off, bbuf);
        // a += calc1(b) + Xp1[s]   (a becomes y0)
        revrnn_halfstep<<<8, 256, 0, stream>>>(bbuf, bw1, bb1, W1b, Xp1 + off, abuf);
    }

    // 6) pack output (B, H) = [y0 | y1]
    revrnn_finalize<<<(B_ * 2 * D_) / 256, 256, 0, stream>>>(abuf, bbuf, out);
}
